// DotProductAtten_77756087927287
// MI455X (gfx1250) — matrix operand order repaired and checked by reading, not compiled, not run on hardware
//
#include <hip/hip_runtime.h>
#include <hip/hip_bf16.h>

typedef __attribute__((ext_vector_type(16))) __bf16 v16bf;
typedef __attribute__((ext_vector_type(8)))  __bf16 v8bf;
typedef __attribute__((ext_vector_type(8)))  float  v8f;
typedef __attribute__((ext_vector_type(8)))  float  f32x8;

// Problem constants (from reference): B=32, Lq=Lk=2048, d=dv=64.
constexpr int kBatch   = 32;
constexpr int kL       = 2048;
constexpr int kD       = 64;
constexpr int kBQ      = 128;   // q rows per block (2 row-tiles of 16 per wave)
constexpr int kRT      = 2;     // row-tiles per wave
constexpr int kBK      = 64;    // k rows per step
constexpr int kWaves   = 4;
constexpr int kThreads = kWaves * 32;
constexpr int kPad     = 72;    // bf16 row stride (36 words -> conflict-free 16-lane gathers)

// ---- fast f32 -> bf16 (round-to-nearest via +0x8000, then take high 16) ----
__device__ inline unsigned pack2_bf16(float a, float b) {
  union { float f; unsigned u; } xa, xb; xa.f = a; xb.f = b;
  unsigned ua = xa.u + 0x8000u;
  unsigned ub = xb.u + 0x8000u;
#if __has_builtin(__builtin_amdgcn_perm)
  // result = { lo16 = a[31:16], hi16 = b[31:16] } in ONE v_perm_b32
  return __builtin_amdgcn_perm(ua, ub, 0x03020706u);
#else
  return (ua >> 16) | (ub & 0xFFFF0000u);
#endif
}
__device__ inline __bf16 to_bf16f(float f) {
  union { float f; unsigned u; } x; x.f = f;
  unsigned short h = (unsigned short)((x.u + 0x8000u) >> 16);
  __bf16 b; __builtin_memcpy(&b, &h, 2); return b;
}
__device__ inline float fast_exp2(float x) {
#if __has_builtin(__builtin_amdgcn_exp2f)
  return __builtin_amdgcn_exp2f(x);
#else
  return exp2f(x);
#endif
}

// XOR-butterfly shuffle within each 16-lane group via v_permlane16_b32 (pure
// VALU, no DScnt wait; both wave32 halves permute independently with the same
// lane-select — exactly the per-half row reduction we need).
template <int MASK>
__device__ inline float xorshuf16(float v) {
#if __has_builtin(__builtin_amdgcn_permlane16)
  unsigned s0, s1;
  if      (MASK == 1) { s0 = 0x67452301u; s1 = 0xEFCDAB89u; }
  else if (MASK == 2) { s0 = 0x54761032u; s1 = 0xDCFE98BAu; }
  else if (MASK == 4) { s0 = 0x32107654u; s1 = 0xBA98FEDCu; }
  else                { s0 = 0xFEDCBA98u; s1 = 0x76543210u; }
  int iv = __builtin_bit_cast(int, v);
  int r  = __builtin_amdgcn_permlane16(iv, iv, (int)s0, (int)s1, false, false);
  return __builtin_bit_cast(float, r);
#else
  return __shfl_xor(v, MASK, 32);
#endif
}

// assemble a 16-element bf16 fragment from two contiguous 16B LDS chunks
__device__ inline v16bf load_frag16(const __bf16* p) {
  v8bf lo = *(const v8bf*)p;
  v8bf hi = *(const v8bf*)(p + 8);
  return __builtin_shufflevector(lo, hi, 0, 1, 2, 3, 4, 5, 6, 7,
                                 8, 9, 10, 11, 12, 13, 14, 15);
}

__global__ __launch_bounds__(kThreads)
void DotProductAtten_77756087927287_kernel(const float* __restrict__ q,
                                           const float* __restrict__ k,
                                           const float* __restrict__ v,
                                           float* __restrict__ out) {
  __shared__ __align__(16) __bf16 Kt[kBK][kPad];               // K tile, row-major (krow, d)
  __shared__ __align__(16) __bf16 Vt[kD][kPad];                // V tile, transposed (dv, krow)
  __shared__ __align__(16) __bf16 Pb[kWaves][16 * kRT][kPad];  // per-wave P, row-major bf16

  const int tid  = threadIdx.x;
  const int wave = tid >> 5;
  const int lane = tid & 31;
  const int half = lane >> 4;   // lane group 0-15 vs 16-31
  const int l16  = lane & 15;

  const int bidx  = blockIdx.x;
  const int b     = bidx / (kL / kBQ);
  const int qt    = bidx % (kL / kBQ);
  const int qbase = qt * kBQ;

  // scores kept in log2 domain: fold 1/sqrt(d) * log2(e) into Q
  const float qscale = 0.125f * 1.44269504088896340736f;

  // ---- Load Q fragments once (A-matrix, 16x32 bf16; 2 frags x 2 row-tiles).
  // Lane layout (contiguous k order used here, same as B; 16-bit A 16x32): row = l16;
  //   elems 0..7  = K = 32f + 16*half + {0..7}
  //   elems 8..15 = K = 32f + 16*half + 8 + {0..7}
  v16bf qa[kRT][2];
#pragma unroll
  for (int rt = 0; rt < kRT; ++rt) {
    const size_t qrow_off =
        ((size_t)b * kL + qbase + wave * (16 * kRT) + rt * 16 + l16) * (size_t)kD;
#pragma unroll
    for (int f = 0; f < 2; ++f) {
      int k0 = 32 * f + 16 * half;
      f32x8 lo = *(const f32x8*)(q + qrow_off + k0);
      f32x8 hi = *(const f32x8*)(q + qrow_off + k0 + 8);
#pragma unroll
      for (int e = 0; e < 8; ++e) {
        qa[rt][f][e]     = to_bf16f(lo[e] * qscale);
        qa[rt][f][e + 8] = to_bf16f(hi[e] * qscale);
      }
    }
  }

  // constant all-ones bf16 B fragment (for row-sums via WMMA)
  v16bf ones;
  {
    unsigned short oneb = 0x3F80;  // bf16(1.0)
    __bf16 o; __builtin_memcpy(&o, &oneb, 2);
#pragma unroll
    for (int e = 0; e < 16; ++e) ones[e] = o;
  }

  // Online-softmax state. Row r = 8*half + i (replicated across the 16-lane half).
  v8f acc[kRT][4];
  const v8f vzero = {0.f, 0.f, 0.f, 0.f, 0.f, 0.f, 0.f, 0.f};
#pragma unroll
  for (int rt = 0; rt < kRT; ++rt)
#pragma unroll
    for (int n = 0; n < 4; ++n) acc[rt][n] = vzero;
  float mrow[kRT][8], lrow[kRT][8];
#pragma unroll
  for (int rt = 0; rt < kRT; ++rt)
#pragma unroll
    for (int i = 0; i < 8; ++i) { mrow[rt][i] = -1e30f; lrow[rt][i] = 0.f; }

  for (int kt = 0; kt < kL / kBK; ++kt) {
    const float* kg = k + ((size_t)b * kL + (size_t)kt * kBK) * kD;
    const float* vg = v + ((size_t)b * kL + (size_t)kt * kBK) * kD;

    __syncthreads();
    // Stage K (row-major bf16) and V (transposed bf16): converted ONCE here via
    // v_perm packing; amortized over 128 q rows of WMMA work.
#pragma unroll
    for (int it = 0; it < (kBK * kD / 4) / kThreads; ++it) {
      int i = tid + it * kThreads;
      int r = (i * 4) / kD, c = (i * 4) % kD;
      float4 kk4 = *(const float4*)(kg + r * kD + c);
      uint2 kp;
      kp.x = pack2_bf16(kk4.x, kk4.y);
      kp.y = pack2_bf16(kk4.z, kk4.w);
      *(uint2*)&Kt[r][c] = kp;                      // 8B DS store
      float4 vv4 = *(const float4*)(vg + r * kD + c);
      Vt[c + 0][r] = to_bf16f(vv4.x);               // transpose: strided b16 stores
      Vt[c + 1][r] = to_bf16f(vv4.y);
      Vt[c + 2][r] = to_bf16f(vv4.z);
      Vt[c + 3][r] = to_bf16f(vv4.w);
    }
    __syncthreads();

    if (kt + 1 < kL / kBK) {  // gfx1250 global_prefetch_b8 for next tiles
      __builtin_prefetch(kg + (size_t)kBK * kD, 0, 1);
      __builtin_prefetch(vg + (size_t)kBK * kD, 0, 1);
    }

    // ---- Phase 1: scores for BOTH row-tiles, sharing each K fragment.
    // Each loaded fragment feeds 2 WMMAs (one per row-tile).
    v8f s[kRT][4];
#pragma unroll
    for (int j = 0; j < 4; ++j) {
      // B frag (bf16 32x16): lane col n = l16 -> krow j*16+l16;
      // elems 0..15 = K(d-index) = 32f + 16*half + {0..15} (contiguous in Kt row)
      v16bf kb0 = load_frag16(&Kt[j * 16 + l16][16 * half]);
      v16bf kb1 = load_frag16(&Kt[j * 16 + l16][32 + 16 * half]);
#pragma unroll
      for (int rt = 0; rt < kRT; ++rt) {
        v8f c0 = __builtin_amdgcn_wmma_f32_16x16x32_bf16(false, qa[rt][0], false, kb0,
                                                         (short)0, vzero, false, false);
        c0 = __builtin_amdgcn_wmma_f32_16x16x32_bf16(false, qa[rt][1], false, kb1,
                                                     (short)0, c0, false, false);
        s[rt][j] = c0;
      }
    }

    // ---- Phase 2: online softmax + P staging per row-tile
    v16bf pa[kRT][2];
#pragma unroll
    for (int rt = 0; rt < kRT; ++rt) {
      float alpha[8];
#pragma unroll
      for (int i = 0; i < 8; ++i) {
        float vmx = fmaxf(fmaxf(s[rt][0][i], s[rt][1][i]),
                          fmaxf(s[rt][2][i], s[rt][3][i]));
        vmx = fmaxf(vmx, xorshuf16<1>(vmx));
        vmx = fmaxf(vmx, xorshuf16<2>(vmx));
        vmx = fmaxf(vmx, xorshuf16<4>(vmx));
        vmx = fmaxf(vmx, xorshuf16<8>(vmx));
        float mn = fmaxf(mrow[rt][i], vmx);
        alpha[i]    = fast_exp2(mrow[rt][i] - mn);   // log2 domain
        mrow[rt][i] = mn;
      }

      // p = exp2(s - m); C-layout -> wave-private LDS (row-major bf16).
      // DS ops are in-order within a wave, no barrier needed for the re-load.
#pragma unroll
      for (int j = 0; j < 4; ++j)
#pragma unroll
        for (int i = 0; i < 8; ++i)
          Pb[wave][rt * 16 + 8 * half + i][j * 16 + l16] =
              to_bf16f(fast_exp2(s[rt][j][i] - mrow[rt][i]));

#pragma unroll
      for (int f = 0; f < 2; ++f) {
        int c0 = 32 * f + 16 * half;
        v8bf lo = *(const v8bf*)&Pb[wave][rt * 16 + l16][c0];
        v8bf hi = *(const v8bf*)&Pb[wave][rt * 16 + l16][c0 + 8];
        pa[rt][f] = __builtin_shufflevector(lo, hi, 0, 1, 2, 3, 4, 5, 6, 7,
                                            8, 9, 10, 11, 12, 13, 14, 15);
      }

      // row sums via WMMA against all-ones B: rs[i] = sum_k P[row_i][k]
      v8f rs = __builtin_amdgcn_wmma_f32_16x16x32_bf16(false, pa[rt][0], false, ones,
                                                       (short)0, vzero, false, false);
      rs = __builtin_amdgcn_wmma_f32_16x16x32_bf16(false, pa[rt][1], false, ones,
                                                   (short)0, rs, false, false);
#pragma unroll
      for (int i = 0; i < 8; ++i) {
        lrow[rt][i] = lrow[rt][i] * alpha[i] + rs[i];
#pragma unroll
        for (int n = 0; n < 4; ++n) acc[rt][n][i] *= alpha[i];
      }
    }

    // ---- Phase 3: O += P @ V, sharing each V fragment across row-tiles
#pragma unroll
    for (int n = 0; n < 4; ++n) {
      // B frag of V: lane col = dv = n*16+l16; K(krow) = 32f + 16*half + {0..15}
      v16bf vb0 = load_frag16(&Vt[n * 16 + l16][16 * half]);
      v16bf vb1 = load_frag16(&Vt[n * 16 + l16][32 + 16 * half]);
#pragma unroll
      for (int rt = 0; rt < kRT; ++rt) {
        acc[rt][n] = __builtin_amdgcn_wmma_f32_16x16x32_bf16(
            false, pa[rt][0], false, vb0, (short)0, acc[rt][n], false, false);
        acc[rt][n] = __builtin_amdgcn_wmma_f32_16x16x32_bf16(
            false, pa[rt][1], false, vb1, (short)0, acc[rt][n], false, false);
      }
    }
  }

  // ---- epilogue: O = acc / l
#pragma unroll
  for (int rt = 0; rt < kRT; ++rt) {
    float rinv[8];
#pragma unroll
    for (int i = 0; i < 8; ++i) rinv[i] = 1.0f / lrow[rt][i];
#pragma unroll
    for (int n = 0; n < 4; ++n)
#pragma unroll
      for (int i = 0; i < 8; ++i) {
        int row = rt * 16 + 8 * half + i;
        size_t o = ((size_t)b * kL + qbase + wave * (16 * kRT) + row) * (size_t)kD +
                   n * 16 + l16;
        out[o] = acc[rt][n][i] * rinv[i];
      }
  }
}

extern "C" void kernel_launch(void* const* d_in, const int* in_sizes, int n_in,
                              void* d_out, int out_size, void* d_ws, size_t ws_size,
                              hipStream_t stream) {
  const float* q = (const float*)d_in[0];
  const float* k = (const float*)d_in[1];
  const float* v = (const float*)d_in[2];
  float* out = (float*)d_out;
  dim3 grid(kBatch * (kL / kBQ));
  DotProductAtten_77756087927287_kernel<<<grid, kThreads, 0, stream>>>(q, k, v, out);
}
